// AttentionPairBias_61280593380098
// MI455X (gfx1250) — matrix operand order repaired and checked by reading, not compiled, not run on hardware
//
#include <hip/hip_runtime.h>
#include <hip/hip_bf16.h>

// ---------------------------------------------------------------------------
// AttentionPairBias for MI455X (gfx1250, wave32, WMMA f16 16x16x32)
// N=1024, C=768, CZ=128, H=16, D=48 (padded to 64 for the QK^T contraction)
// ---------------------------------------------------------------------------

typedef __attribute__((ext_vector_type(16))) _Float16 v16h;
typedef __attribute__((ext_vector_type(8)))  _Float16 v8h;
typedef __attribute__((ext_vector_type(8)))  float    v8f;

#define WMMA16(A_, B_, C_) \
    __builtin_amdgcn_wmma_f32_16x16x32_f16(false, (A_), false, (B_), (short)0, (C_), false, false)

// A-fragment loader (16x32 f16, M x K), contiguous K order:
// lane<16 : K in [0,16) of row lane
// lane>=16: K in [16,32) of row lane-16
// base = &row[k0]; hf = lane>>4
__device__ __forceinline__ v16h load_a16(const _Float16* base, int hf) {
    return *(const v16h*)(base + hf * 16);
}

static constexpr int   kN  = 1024;
static constexpr int   kC  = 768;
static constexpr int   kCZ = 128;
static constexpr int   kH  = 16;
static constexpr int   kD  = 48;
static constexpr int   kDP = 64;          // padded head dim
static constexpr float kQScale = 0.14433756729740643f;  // 1/sqrt(48)

// ---------------------------------------------------------------------------
// Weight conversion: wq/wk/wv/wg [C][H*D] f32 -> wpadT [4][H*64][C] f16
// (transposed, head-dim zero-padded to 64; q pre-scaled by 1/sqrt(D))
// ---------------------------------------------------------------------------
__global__ void k_cvt_qkvg(const float* __restrict__ wq, const float* __restrict__ wk,
                           const float* __restrict__ wv, const float* __restrict__ wg,
                           _Float16* __restrict__ wpadT) {
    size_t idx = (size_t)blockIdx.x * 256 + threadIdx.x;    // 4*1024*768 total
    int which = (int)(idx / (size_t)(kH * kDP * kC));
    size_t rem = idx % (size_t)(kH * kDP * kC);
    int j = (int)(rem / kC);                                // padded col 0..1023
    int k = (int)(rem % kC);
    int h = j >> 6, d = j & 63;
    float v = 0.f;
    if (d < kD) {
        const float* w = (which == 0) ? wq : (which == 1) ? wk : (which == 2) ? wv : wg;
        v = w[(size_t)k * (kH * kD) + h * kD + d];
        if (which == 0) v *= kQScale;
    }
    wpadT[idx] = (_Float16)v;
}

// wo [H*D][C] -> woT [C][H*D] f16
__global__ void k_cvt_wo(const float* __restrict__ wo, _Float16* __restrict__ woT) {
    size_t idx = (size_t)blockIdx.x * 256 + threadIdx.x;    // 768*768
    int c  = (int)(idx / kC);
    int hd = (int)(idx % kC);
    woT[idx] = (_Float16)wo[(size_t)hd * kC + c];
}

// w_z [CZ][H] -> wzT [H][CZ] f16 with LN-z scale folded; bz[h] = sum_c ln_b[c]*w_z[c][h]
__global__ void k_cvt_wz(const float* __restrict__ wz, const float* __restrict__ lnw,
                         const float* __restrict__ lnb, _Float16* __restrict__ wzT,
                         float* __restrict__ bz) {
    int t = threadIdx.x;
    for (int i = t; i < kH * kCZ; i += 256) {
        int h = i >> 7, c = i & 127;
        wzT[i] = (_Float16)(lnw[c] * wz[(size_t)c * kH + h]);
    }
    if (t < kH) {
        float s = 0.f;
        for (int c = 0; c < kCZ; c++) s += lnb[c] * wz[(size_t)c * kH + t];
        bz[t] = s;
    }
}

// ---------------------------------------------------------------------------
// LayerNorm of a: [N][C] f32 -> f16. One block (256 threads) per row.
// ---------------------------------------------------------------------------
__global__ void k_ln_a(const float* __restrict__ a, const float* __restrict__ w,
                       const float* __restrict__ b, _Float16* __restrict__ an) {
    int n = blockIdx.x, t = threadIdx.x;
    const float* row = a + (size_t)n * kC;
    float x0 = row[t], x1 = row[t + 256], x2 = row[t + 512];
    float s = x0 + x1 + x2;
    for (int m = 16; m >= 1; m >>= 1) s += __shfl_xor(s, m, 32);
    __shared__ float red[8], red2[8];
    int wv = t >> 5, ln = t & 31;
    if (ln == 0) red[wv] = s;
    __syncthreads();
    float tot = 0.f;
    for (int i = 0; i < 8; i++) tot += red[i];
    float mu = tot * (1.f / kC);
    float d0 = x0 - mu, d1 = x1 - mu, d2 = x2 - mu;
    float s2 = d0 * d0 + d1 * d1 + d2 * d2;
    for (int m = 16; m >= 1; m >>= 1) s2 += __shfl_xor(s2, m, 32);
    if (ln == 0) red2[wv] = s2;
    __syncthreads();
    float tv = 0.f;
    for (int i = 0; i < 8; i++) tv += red2[i];
    float rs = rsqrtf(tv * (1.f / kC) + 1e-5f);
    an[(size_t)n * kC + t]       = (_Float16)(d0 * rs * w[t]       + b[t]);
    an[(size_t)n * kC + t + 256] = (_Float16)(d1 * rs * w[t + 256] + b[t + 256]);
    an[(size_t)n * kC + t + 512] = (_Float16)(d2 * rs * w[t + 512] + b[t + 512]);
}

// ---------------------------------------------------------------------------
// Fused QKVG projection GEMM: an[N][768] x wpadT[which][1024][768]
// which: 0->q_pad f16 [N][1024], 1->k_pad f16 [N][1024],
//        2->v_t  f16 [H*48][N] (transposed for the PV B-operand),
//        3->g    f32 [N][768] = sigmoid(.+bg)
// block 256 (8 waves, 16 rows each), grid (N/128, 1024/64, 4). K-loop 768/32.
// ---------------------------------------------------------------------------
__global__ void k_proj(const _Float16* __restrict__ an, const _Float16* __restrict__ wpadT,
                       const float* __restrict__ bg, _Float16* __restrict__ qp,
                       _Float16* __restrict__ kp, _Float16* __restrict__ vt,
                       float* __restrict__ g) {
    int which = blockIdx.z;
    const _Float16* B = wpadT + (size_t)which * (kH * kDP) * kC;
    int wave = threadIdx.x >> 5, lane = threadIdx.x & 31;
    int lr = lane & 15, hf = lane >> 4;
    int n0 = blockIdx.x * 128 + wave * 16;
    int j0 = blockIdx.y * 64;
    v8f acc[4] = {{}, {}, {}, {}};
    const _Float16* arow = an + (size_t)(n0 + lr) * kC;
    for (int k0 = 0; k0 < kC; k0 += 32) {
        v16h Aa = load_a16(arow + k0, hf);
        #pragma unroll
        for (int jt = 0; jt < 4; jt++) {
            v16h Bb = *(const v16h*)(B + (size_t)(j0 + jt * 16 + lr) * kC + k0 + hf * 16);
            acc[jt] = WMMA16(Aa, Bb, acc[jt]);
        }
    }
    #pragma unroll
    for (int jt = 0; jt < 4; jt++) {
        #pragma unroll
        for (int r = 0; r < 8; r++) {
            int n = n0 + r + 8 * hf;
            int j = j0 + jt * 16 + lr;
            float v = acc[jt][r];
            if (which == 0) {
                qp[(size_t)n * (kH * kDP) + j] = (_Float16)v;
            } else if (which == 1) {
                kp[(size_t)n * (kH * kDP) + j] = (_Float16)v;
            } else {
                int h = j >> 6, d = j & 63;
                if (d < kD) {
                    if (which == 2) vt[(size_t)(h * kD + d) * kN + n] = (_Float16)v;
                    else {
                        float s = v + bg[h * kD + d];
                        g[(size_t)n * kC + h * kD + d] = 1.f / (1.f + __expf(-s));
                    }
                }
            }
        }
    }
}

// ---------------------------------------------------------------------------
// Pair bias: LN(z) @ (lnw*w_z) + bz  -> pb f16 [H][N][N]
// Each wave handles 16 (i,j) pairs: stage z rows as f16 in LDS, per-row
// mean/rsqrt in registers, 4 WMMAs (K=128) against folded w_z.
// block 256 (8 waves), grid N*N/128 = 8192.
// ---------------------------------------------------------------------------
__global__ void k_pair(const float* __restrict__ z, const _Float16* __restrict__ wzT,
                       const float* __restrict__ bz, _Float16* __restrict__ pb) {
    __shared__ alignas(32) _Float16 zl[8][16 * kCZ];    // 32 KB
    int wave = threadIdx.x >> 5, lane = threadIdx.x & 31;
    int lr = lane & 15, hf = lane >> 4;
    size_t p0 = ((size_t)blockIdx.x * 8 + wave) * 16;
    _Float16* L = zl[wave];
    const float* zr = z + (p0 + lr) * (size_t)kCZ + hf * 64;
    float s1 = 0.f, s2 = 0.f;
    #pragma unroll
    for (int i = 0; i < 64; i += 4) {
        float4 v = *(const float4*)(zr + i);
        s1 += v.x + v.y + v.z + v.w;
        s2 += v.x * v.x + v.y * v.y + v.z * v.z + v.w * v.w;
        int o = lr * kCZ + hf * 64 + i;
        L[o + 0] = (_Float16)v.x; L[o + 1] = (_Float16)v.y;
        L[o + 2] = (_Float16)v.z; L[o + 3] = (_Float16)v.w;
    }
    s1 += __shfl_xor(s1, 16, 32);
    s2 += __shfl_xor(s2, 16, 32);
    float mu = s1 * (1.f / kCZ);
    float rs = rsqrtf(s2 * (1.f / kCZ) - mu * mu + 1e-5f);
    v8f acc = {};
    for (int k0 = 0; k0 < kCZ; k0 += 32) {
        v16h raw = load_a16(L + lr * kCZ + k0, hf);
        v16h Aa;
        #pragma unroll
        for (int i = 0; i < 16; i++) Aa[i] = (_Float16)(((float)raw[i] - mu) * rs);
        v16h Bb = *(const v16h*)(wzT + (size_t)lr * kCZ + k0 + hf * 16);
        acc = WMMA16(Aa, Bb, acc);
    }
    float bzv = bz[lr];                       // D-layout column = head = lr
    #pragma unroll
    for (int r = 0; r < 8; r++) {
        size_t p = p0 + r + 8 * hf;
        pb[(size_t)lr * kN * kN + p] = (_Float16)(acc[r] + bzv);
    }
}

// ---------------------------------------------------------------------------
// Flash attention. One wave = (head h, 16 query rows). block 128 (4 waves),
// grid (H, N/64). Per 32-key tile: 4 WMMAs for QK^T, online softmax in f32,
// P re-layout via LDS, 3 WMMAs for P@V. Gate applied, og f16 [N][H*D].
// ---------------------------------------------------------------------------
__global__ void k_attn(const _Float16* __restrict__ qp, const _Float16* __restrict__ kp,
                       const _Float16* __restrict__ vt, const _Float16* __restrict__ pb,
                       const float* __restrict__ mask, const float* __restrict__ g,
                       _Float16* __restrict__ og) {
    __shared__ alignas(32) _Float16 plds[4][16 * 32];
    int wave = threadIdx.x >> 5, lane = threadIdx.x & 31;
    int lr = lane & 15, hf = lane >> 4;
    int h  = blockIdx.x;
    int q0 = blockIdx.y * 64 + wave * 16;
    const _Float16* qrow = qp + (size_t)(q0 + lr) * (kH * kDP) + h * kDP;
    v16h A0 = load_a16(qrow, hf);        // K = d 0..31
    v16h A1 = load_a16(qrow + 32, hf);   // K = d 32..63 (pad zeros)
    const _Float16* pbh = pb + (size_t)h * kN * kN;
    _Float16* P = plds[wave];
    v8f O0 = {}, O1 = {}, O2 = {};
    float mx[8], li[8];
    #pragma unroll
    for (int r = 0; r < 8; r++) { mx[r] = -1e30f; li[r] = 0.f; }

    for (int m0 = 0; m0 < kN; m0 += 32) {
        v8f lg[2];
        #pragma unroll
        for (int t = 0; t < 2; t++) {
            int mc = m0 + t * 16;
            const _Float16* krow = kp + (size_t)(mc + lr) * (kH * kDP) + h * kDP;
            v16h B0 = *(const v16h*)(krow + hf * 16);
            v16h B1 = *(const v16h*)(krow + 32 + hf * 16);
            v8f a = {};
            a = WMMA16(A0, B0, a);
            a = WMMA16(A1, B1, a);
            float mb = 1e9f * (mask[mc + lr] - 1.0f);   // column bias
            #pragma unroll
            for (int r = 0; r < 8; r++)
                a[r] += (float)pbh[(size_t)(q0 + r + 8 * hf) * kN + mc + lr] + mb;
            lg[t] = a;
        }
        // per-row max across 16 lanes of this half-wave
        float newmx[8];
        #pragma unroll
        for (int r = 0; r < 8; r++) {
            float m = fmaxf(lg[0][r], lg[1][r]);
            for (int s = 8; s >= 1; s >>= 1) m = fmaxf(m, __shfl_xor(m, s, 16));
            newmx[r] = fmaxf(mx[r], m);
        }
        float psum[8];
        #pragma unroll
        for (int r = 0; r < 8; r++) {
            float sc = __expf(mx[r] - newmx[r]);
            O0[r] *= sc; O1[r] *= sc; O2[r] *= sc;
            li[r] *= sc;
            float e0 = __expf(lg[0][r] - newmx[r]);
            float e1 = __expf(lg[1][r] - newmx[r]);
            P[(r + 8 * hf) * 32 + lr]      = (_Float16)e0;
            P[(r + 8 * hf) * 32 + 16 + lr] = (_Float16)e1;
            psum[r] = e0 + e1;
            mx[r] = newmx[r];
        }
        #pragma unroll
        for (int r = 0; r < 8; r++) {
            float s = psum[r];
            for (int st = 8; st >= 1; st >>= 1) s += __shfl_xor(s, st, 16);
            li[r] += s;
        }
        // P (16x32) as A; V tiles as B (K = key index, N-col = d)
        v16h Pa = load_a16(P + lr * 32, hf);
        v16h V0 = *(const v16h*)(vt + (size_t)(h * kD +  0 + lr) * kN + m0 + hf * 16);
        v16h V1 = *(const v16h*)(vt + (size_t)(h * kD + 16 + lr) * kN + m0 + hf * 16);
        v16h V2 = *(const v16h*)(vt + (size_t)(h * kD + 32 + lr) * kN + m0 + hf * 16);
        O0 = WMMA16(Pa, V0, O0);
        O1 = WMMA16(Pa, V1, O1);
        O2 = WMMA16(Pa, V2, O2);
    }
    #pragma unroll
    for (int r = 0; r < 8; r++) {
        int n = q0 + r + 8 * hf;
        float inv = 1.0f / li[r];
        size_t base = (size_t)n * kC + h * kD;
        og[base +  0 + lr] = (_Float16)(O0[r] * inv * g[base +  0 + lr]);
        og[base + 16 + lr] = (_Float16)(O1[r] * inv * g[base + 16 + lr]);
        og[base + 32 + lr] = (_Float16)(O2[r] * inv * g[base + 32 + lr]);
    }
}

// ---------------------------------------------------------------------------
// Output GEMM: out[N][C] f32 = og[N][768] @ woT^T + bo
// block 256 (8 waves, 16 rows each), grid (N/128, C/64).
// ---------------------------------------------------------------------------
__global__ void k_out(const _Float16* __restrict__ og, const _Float16* __restrict__ woT,
                      const float* __restrict__ bo, float* __restrict__ out) {
    int wave = threadIdx.x >> 5, lane = threadIdx.x & 31;
    int lr = lane & 15, hf = lane >> 4;
    int n0 = blockIdx.x * 128 + wave * 16;
    int j0 = blockIdx.y * 64;
    v8f acc[4] = {{}, {}, {}, {}};
    const _Float16* arow = og + (size_t)(n0 + lr) * kC;
    for (int k0 = 0; k0 < kC; k0 += 32) {
        v16h Aa = load_a16(arow + k0, hf);
        #pragma unroll
        for (int jt = 0; jt < 4; jt++) {
            v16h Bb = *(const v16h*)(woT + (size_t)(j0 + jt * 16 + lr) * kC + k0 + hf * 16);
            acc[jt] = WMMA16(Aa, Bb, acc[jt]);
        }
    }
    #pragma unroll
    for (int jt = 0; jt < 4; jt++)
        #pragma unroll
        for (int r = 0; r < 8; r++) {
            int n = n0 + r + 8 * hf;
            int j = j0 + jt * 16 + lr;
            out[(size_t)n * kC + j] = acc[jt][r] + bo[j];
        }
}

// ---------------------------------------------------------------------------
extern "C" void kernel_launch(void* const* d_in, const int* in_sizes, int n_in,
                              void* d_out, int out_size, void* d_ws, size_t ws_size,
                              hipStream_t stream) {
    (void)in_sizes; (void)n_in; (void)out_size; (void)ws_size;
    const float* a      = (const float*)d_in[0];
    const float* z      = (const float*)d_in[1];
    const float* mask   = (const float*)d_in[2];
    const float* ln_a_w = (const float*)d_in[3];
    const float* ln_a_b = (const float*)d_in[4];
    const float* ln_z_w = (const float*)d_in[5];
    const float* ln_z_b = (const float*)d_in[6];
    const float* w_z    = (const float*)d_in[7];
    const float* wq     = (const float*)d_in[8];
    const float* wk     = (const float*)d_in[9];
    const float* wv     = (const float*)d_in[10];
    const float* wg     = (const float*)d_in[11];
    const float* bg     = (const float*)d_in[12];
    const float* wo     = (const float*)d_in[13];
    const float* bo     = (const float*)d_in[14];
    float* out = (float*)d_out;

    char* ws = (char*)d_ws;
    auto carve = [&](size_t bytes) -> char* {
        char* p = ws;
        ws += (bytes + 255) & ~(size_t)255;
        return p;
    };
    _Float16* wpadT = (_Float16*)carve((size_t)4 * kH * kDP * kC * 2);   // 6.0 MB
    _Float16* woT   = (_Float16*)carve((size_t)kC * kC * 2);             // 1.1 MB
    _Float16* wzT   = (_Float16*)carve((size_t)kH * kCZ * 2);
    float*    bz    = (float*)   carve((size_t)kH * 4);
    _Float16* an    = (_Float16*)carve((size_t)kN * kC * 2);             // 1.5 MB
    _Float16* qp    = (_Float16*)carve((size_t)kN * kH * kDP * 2);       // 2.0 MB
    _Float16* kp    = (_Float16*)carve((size_t)kN * kH * kDP * 2);       // 2.0 MB
    _Float16* vt    = (_Float16*)carve((size_t)kH * kD * kN * 2);        // 1.5 MB
    float*    gbuf  = (float*)   carve((size_t)kN * kC * 4);             // 3.0 MB
    _Float16* og    = (_Float16*)carve((size_t)kN * kC * 2);             // 1.5 MB
    _Float16* pb    = (_Float16*)carve((size_t)kH * kN * kN * 2);        // 32 MB

    k_cvt_qkvg<<<(4 * kH * kDP * kC) / 256, 256, 0, stream>>>(wq, wk, wv, wg, wpadT);
    k_cvt_wo  <<<(kC * kC) / 256, 256, 0, stream>>>(wo, woT);
    k_cvt_wz  <<<1, 256, 0, stream>>>(w_z, ln_z_w, ln_z_b, wzT, bz);
    k_ln_a    <<<kN, 256, 0, stream>>>(a, ln_a_w, ln_a_b, an);
    k_proj    <<<dim3(kN / 128, (kH * kDP) / 64, 4), 256, 0, stream>>>(an, wpadT, bg, qp, kp, vt, gbuf);
    k_pair    <<<(kN * kN) / 128, 256, 0, stream>>>(z, wzT, bz, pb);
    k_attn    <<<dim3(kH, kN / 64), 128, 0, stream>>>(qp, kp, vt, pb, mask, gbuf, og);
    k_out     <<<dim3(kN / 128, kC / 64), 256, 0, stream>>>(og, woT, bo, out);
}
